// FA2VarlenPipeline_15358803051127
// MI455X (gfx1250) — matrix operand order repaired and checked by reading, not compiled, not run on hardware
//
#include <hip/hip_runtime.h>
#include <math.h>

// ---------------------------------------------------------------------------
// ViT-B/16 forward with token pruning, CDNA5 (gfx1250) WMMA bf16 pipeline.
// Round 2: attention now stages Q/K/V^T into LDS with
// GLOBAL_LOAD_ASYNC_TO_LDS_B128 (+ s_wait_asynccnt), and does softmax in place
// inside the LDS logits buffer so everything fits in the 320KB WGP LDS.
// ---------------------------------------------------------------------------

typedef __bf16 bf16;
typedef __attribute__((ext_vector_type(16))) __bf16 v16bf;
typedef __attribute__((ext_vector_type(8)))  __bf16 v8bf;
typedef __attribute__((ext_vector_type(8)))  float  v8f;

#define B_     64
#define S_     197
#define D_     768
#define H_     12
#define HD_    64
#define TD_    2304
#define MLP_   3072
#define L_     12
#define NKEEP_ 98
#define QP_    208   // padded query rows  (multiple of 16)
#define SP_    224   // padded key length  (multiple of 32)
#define MTOK_  (B_*S_)     // 12608
#define MPATCH_ (B_*196)   // 12544
#define BH_    (B_*H_)     // 768

// LDS layout for the fused attention kernel (bytes)
#define LG_BYTES  ((size_t)QP_ * SP_ * 4)        // 186368: f32 logits / bf16 attn (aliased)
#define QS_BYTES  ((size_t)QP_ * HD_ * 2)        // 26624
#define KS_BYTES  ((size_t)SP_ * HD_ * 2)        // 28672
#define VTS_BYTES ((size_t)HD_ * SP_ * 2)        // 28672
#define ATTN_LDS  (LG_BYTES + QS_BYTES + KS_BYTES + VTS_BYTES)  // 270336 < 320KB

// ---------------------------------------------------------------------------
// WMMA helpers (A and B both use contiguous k order: lane half h holds k = 16h + s)
// ---------------------------------------------------------------------------
static __device__ inline v8f wmma_bf16(v16bf a, v16bf b, v8f c) {
  return __builtin_amdgcn_wmma_f32_16x16x32_bf16(false, a, false, b, (short)0, c,
                                                 false, false);
}

// A-matrix 16x32 bf16 fragment. lanes 0-15: rows M=0..15 with K = k0+0..15
// lanes 16-31: same rows with K = k0+16..31. Two 16B chunks per lane.
static __device__ inline v16bf load_a_frag(const bf16* base, int ld, int row0,
                                           int k0, int lane, int rowmax) {
  int r = row0 + (lane & 15);
  if (r > rowmax) r = rowmax;
  const bf16* p = base + (size_t)r * ld + k0 + ((lane >> 4) << 4);
  v8bf c0 = *(const v8bf*)p;
  v8bf c1 = *(const v8bf*)(p + 8);
  return __builtin_shufflevector(c0, c1, 0, 1, 2, 3, 4, 5, 6, 7,
                                 8, 9, 10, 11, 12, 13, 14, 15);
}

// B-matrix 32x16 bf16 fragment from W[N,K] row-major (C = A * W^T form).
// lane n = col; lanes 0-15 hold K=k0+0..15, lanes 16-31 hold K=k0+16..31;
// one contiguous 32B chunk per lane.
static __device__ inline v16bf load_b_frag(const bf16* base, int ld, int col0,
                                           int k0, int lane, int colmax) {
  int n = col0 + (lane & 15);
  if (n > colmax) n = colmax;
  const bf16* p = base + (size_t)n * ld + k0 + ((lane >> 4) << 4);
  return *(const v16bf*)p;
}

static __device__ inline float gelu_exact(float v) {
  return 0.5f * v * (1.0f + erff(v * 0.70710678118654752f));
}

// ---------------------------------------------------------------------------
// Async global->LDS copy (CDNA5 GLOBAL_LOAD_ASYNC_TO_LDS_B128, ASYNCcnt).
// Per-lane: 16 bytes from the lane's global address to the lane's LDS address.
// Generic shared pointers carry the LDS offset in their low 32 bits.
// ---------------------------------------------------------------------------
static __device__ inline void async_ld16(unsigned lds_addr, const void* gaddr) {
  asm volatile("global_load_async_to_lds_b128 %0, %1, off"
               :: "v"(lds_addr), "v"(gaddr)
               : "memory");
}

static __device__ inline void stage_async(void* dst, const void* src, int bytes,
                                          int tid) {
  const char* s = (const char*)src;
  const unsigned d = (unsigned)(unsigned long long)dst;
  for (int ofs = tid * 16; ofs < bytes; ofs += 256 * 16)
    async_ld16(d + (unsigned)ofs, s + ofs);
}

static __device__ inline void wait_async0() {
  asm volatile("s_wait_asynccnt 0x0" ::: "memory");
}

// ---------------------------------------------------------------------------
// Generic GEMM: C[M,N] = A[M,K] * W[N,K]^T (+bias)(GELU)(+resid) -> f32/bf16
// 128 threads = 4 waves in 2x2; each wave computes a 64x64 tile (4x4 WMMA).
// ---------------------------------------------------------------------------
__global__ void __launch_bounds__(128)
gemm_bf16_kernel(const bf16* __restrict__ A, const bf16* __restrict__ W,
                 int M, int N, int K,
                 const float* __restrict__ bias,
                 const float* __restrict__ resid,
                 float* __restrict__ outF, bf16* __restrict__ outB,
                 int gelu_flag) {
  const int tid  = threadIdx.x;
  const int lane = tid & 31;
  const int wave = tid >> 5;
  const int m0 = blockIdx.y * 128 + (wave >> 1) * 64;
  const int n0 = blockIdx.x * 128 + (wave & 1) * 64;

  v8f acc[4][4];
  const v8f vz = {0.f, 0.f, 0.f, 0.f, 0.f, 0.f, 0.f, 0.f};
#pragma unroll
  for (int i = 0; i < 4; ++i)
#pragma unroll
    for (int j = 0; j < 4; ++j) acc[i][j] = vz;

  for (int k0 = 0; k0 < K; k0 += 32) {
    v16bf af[4], bf_[4];
#pragma unroll
    for (int i = 0; i < 4; ++i)
      af[i] = load_a_frag(A, K, m0 + i * 16, k0, lane, M - 1);
#pragma unroll
    for (int j = 0; j < 4; ++j)
      bf_[j] = load_b_frag(W, K, n0 + j * 16, k0, lane, N - 1);
#pragma unroll
    for (int i = 0; i < 4; ++i)
#pragma unroll
      for (int j = 0; j < 4; ++j)
        acc[i][j] = wmma_bf16(af[i], bf_[j], acc[i][j]);
  }

  // Epilogue. C layout: lane -> col n0+(lane&15); vgpr r -> row r + 8*(lane>>4).
  const int cn = lane & 15, half = lane >> 4;
#pragma unroll
  for (int j = 0; j < 4; ++j) {
    const int n = n0 + j * 16 + cn;
    if (n >= N) continue;
    const float bv = bias ? bias[n] : 0.f;
#pragma unroll
    for (int i = 0; i < 4; ++i) {
#pragma unroll
      for (int r = 0; r < 8; ++r) {
        const int m = m0 + i * 16 + r + 8 * half;
        if (m < M) {
          float v = acc[i][j][r] + bv;
          if (gelu_flag) v = gelu_exact(v);
          const size_t idx = (size_t)m * N + n;
          if (resid) v += resid[idx];
          if (outF) outF[idx] = v;
          if (outB) outB[idx] = (bf16)v;
        }
      }
    }
  }
}

// ---------------------------------------------------------------------------
// Fused attention for one (b,h). Q/K/V^T async-staged into LDS, logits (WMMA)
// into LDS f32, masked softmax IN PLACE (bf16 attn row overwrites the front
// half of its own f32 logit row; per-row ownership + read-before-write makes
// this safe), then attn @ V^T with WMMA A/B fragments read from LDS.
// ---------------------------------------------------------------------------
__global__ void __launch_bounds__(256)
attention_kernel(const bf16* __restrict__ q, const bf16* __restrict__ k,
                 const bf16* __restrict__ vt,
                 const unsigned char* __restrict__ mask,
                 bf16* __restrict__ o) {
  extern __shared__ char smem[];
  float* lg  = (float*)smem;                                   // [QP_][SP_] f32
  bf16*  att = (bf16*)smem;                                    // rows stride 2*SP_
  bf16*  qs  = (bf16*)(smem + LG_BYTES);                       // [QP_][HD_]
  bf16*  ks  = (bf16*)(smem + LG_BYTES + QS_BYTES);            // [SP_][HD_]
  bf16*  vts = (bf16*)(smem + LG_BYTES + QS_BYTES + KS_BYTES); // [HD_][SP_]

  const int bh = blockIdx.x;
  const int b = bh / H_, h = bh % H_;
  const bf16* Q  = q  + (size_t)bh * QP_ * HD_;
  const bf16* Kp = k  + (size_t)bh * SP_ * HD_;
  const bf16* VT = vt + (size_t)bh * HD_ * SP_;
  const int tid = threadIdx.x;
  const int lane = tid & 31, wave = tid >> 5;
  const int cn = lane & 15, half = lane >> 4;
  const v8f vz = {0.f, 0.f, 0.f, 0.f, 0.f, 0.f, 0.f, 0.f};

  // ---- Stage Q, K, V^T into LDS via async copies ------------------------
  stage_async(qs,  Q,  (int)QS_BYTES,  tid);
  stage_async(ks,  Kp, (int)KS_BYTES,  tid);
  stage_async(vts, VT, (int)VTS_BYTES, tid);
  wait_async0();
  __syncthreads();

  // ---- Phase 1: logits = (Q * HD^-0.5) @ K^T, tiles 13 x 14 -------------
  for (int t = wave; t < (QP_ / 16) * (SP_ / 16); t += 8) {
    const int qt = t / (SP_ / 16), kt = t % (SP_ / 16);
    v8f acc = vz;
#pragma unroll
    for (int kk = 0; kk < HD_; kk += 32) {
      v16bf a  = load_a_frag(qs, HD_, qt * 16, kk, lane, QP_ - 1);
      v16bf bb = load_b_frag(ks, HD_, kt * 16, kk, lane, SP_ - 1);
      acc = wmma_bf16(a, bb, acc);
    }
#pragma unroll
    for (int r = 0; r < 8; ++r) {
      const int qm = qt * 16 + r + 8 * half;
      lg[(size_t)qm * SP_ + kt * 16 + cn] = acc[r] * 0.125f;  // HD^-0.5
    }
  }
  __syncthreads();

  // ---- Phase 2: masked softmax, one thread per row, in place ------------
  for (int row = tid; row < QP_; row += 256) {
    bf16* arow = att + (size_t)row * (2 * SP_);  // front half of f32 row
    if (row >= S_) {
      for (int c = 0; c < SP_; ++c) arow[c] = (bf16)0.f;
      continue;
    }
    const float* lrow = lg + (size_t)row * SP_;
    float mx = -3.0e38f;
    for (int c = 0; c < S_; ++c) {
      const bool valid = (mask == nullptr) || mask[(size_t)b * S_ + c];
      const float v = valid ? lrow[c] : -1.0e9f;
      mx = fmaxf(mx, v);
    }
    float sum = 0.f;
    for (int c = 0; c < S_; ++c) {
      const bool valid = (mask == nullptr) || mask[(size_t)b * S_ + c];
      sum += valid ? expf(lrow[c] - mx) : 0.f;
    }
    const float inv = 1.f / sum;
    for (int c = 0; c < SP_; ++c) {  // bf16 write c clobbers f32 elem c/2 <= c
      float e = 0.f;
      if (c < S_) {
        const bool valid = (mask == nullptr) || mask[(size_t)b * S_ + c];
        if (valid) e = expf(lrow[c] - mx) * inv;
      }
      arow[c] = (bf16)e;
    }
  }
  __syncthreads();

  // ---- Phase 3: out = attn @ V  (A from LDS attn, B from LDS V^T) -------
  for (int t = wave; t < (QP_ / 16) * (HD_ / 16); t += 8) {
    const int qt = t / (HD_ / 16), nt = t % (HD_ / 16);
    v8f acc = vz;
#pragma unroll
    for (int kk = 0; kk < SP_; kk += 32) {
      v16bf a  = load_a_frag(att, 2 * SP_, qt * 16, kk, lane, QP_ - 1);
      v16bf bb = load_b_frag(vts, SP_, nt * 16, kk, lane, HD_ - 1);
      acc = wmma_bf16(a, bb, acc);
    }
#pragma unroll
    for (int r = 0; r < 8; ++r) {
      const int qm = qt * 16 + r + 8 * half;
      if (qm < S_) {
        const size_t idx = ((size_t)(b * S_ + qm)) * D_ + h * HD_ + nt * 16 + cn;
        o[idx] = (bf16)acc[r];
      }
    }
  }
}

// ---------------------------------------------------------------------------
// Support kernels
// ---------------------------------------------------------------------------
__global__ void cvt_kernel(const float* __restrict__ s, bf16* __restrict__ d,
                           size_t n) {
  size_t i = (size_t)blockIdx.x * blockDim.x + threadIdx.x;
  const size_t step = (size_t)gridDim.x * blockDim.x;
  for (; i < n; i += step) d[i] = (bf16)s[i];
}

// images NCHW (B,3,224,224) -> patches bf16 [B*196, 768], col = c*256+iy*16+ix
__global__ void im2col_kernel(const float* __restrict__ img,
                              bf16* __restrict__ patches) {
  const size_t n = (size_t)MPATCH_ * D_;
  size_t i = (size_t)blockIdx.x * blockDim.x + threadIdx.x;
  const size_t step = (size_t)gridDim.x * blockDim.x;
  for (; i < n; i += step) {
    const int col = (int)(i % D_);
    const size_t r = i / D_;
    const int b = (int)(r / 196), p = (int)(r % 196);
    const int py = p / 14, px = p % 14;
    const int c = col / 256, rem = col % 256;
    const int iy = rem / 16, ix = rem % 16;
    patches[i] = (bf16)img[(((size_t)b * 3 + c) * 224 + py * 16 + iy) * 224 +
                           px * 16 + ix];
  }
}

// x[b,0,:] = cls + pos[0]; x[b,1+p,:] = pe[b,p,:] + pos[1+p]
__global__ void assemble_kernel(const float* __restrict__ pe,
                                const float* __restrict__ cls,
                                const float* __restrict__ pos,
                                float* __restrict__ x) {
  const size_t n = (size_t)MTOK_ * D_;
  size_t i = (size_t)blockIdx.x * blockDim.x + threadIdx.x;
  const size_t step = (size_t)gridDim.x * blockDim.x;
  for (; i < n; i += step) {
    const int d = (int)(i % D_);
    const size_t t = i / D_;
    const int s = (int)(t % S_);
    const int b = (int)(t / S_);
    float v = (s == 0) ? cls[d] : pe[((size_t)b * 196 + (s - 1)) * D_ + d];
    x[i] = v + pos[(size_t)s * D_ + d];
  }
}

// block-per-row layernorm; input rows at x + row*row_stride; bf16 out [rows,768]
__global__ void __launch_bounds__(256)
layernorm_kernel(const float* __restrict__ x, long long row_stride,
                 const float* __restrict__ w, const float* __restrict__ bb,
                 bf16* __restrict__ out, int nrows) {
  __shared__ float red[256];
  const int row = blockIdx.x;
  if (row >= nrows) return;
  const float* xr = x + (size_t)row * row_stride;
  const int tid = threadIdx.x;
  float s = 0.f;
  for (int d = tid; d < D_; d += 256) s += xr[d];
  red[tid] = s;
  __syncthreads();
  for (int st = 128; st > 0; st >>= 1) {
    if (tid < st) red[tid] += red[tid + st];
    __syncthreads();
  }
  const float mean = red[0] / (float)D_;
  __syncthreads();
  float s2 = 0.f;
  for (int d = tid; d < D_; d += 256) {
    const float t = xr[d] - mean;
    s2 += t * t;
  }
  red[tid] = s2;
  __syncthreads();
  for (int st = 128; st > 0; st >>= 1) {
    if (tid < st) red[tid] += red[tid + st];
    __syncthreads();
  }
  const float rstd = rsqrtf(red[0] / (float)D_ + 1e-6f);
  for (int d = tid; d < D_; d += 256)
    out[(size_t)row * D_ + d] = (bf16)((xr[d] - mean) * rstd * w[d] + bb[d]);
}

// qkv f32 [B*S, 3*D] -> q [bh,QP,HD], k [bh,SP,HD], vt [bh,HD,SP] (zero pad)
__global__ void repack_kernel(const float* __restrict__ qkv,
                              bf16* __restrict__ q, bf16* __restrict__ k,
                              bf16* __restrict__ vt) {
  const size_t n = (size_t)BH_ * SP_ * HD_;
  size_t i = (size_t)blockIdx.x * blockDim.x + threadIdx.x;
  const size_t step = (size_t)gridDim.x * blockDim.x;
  for (; i < n; i += step) {
    const int hd = (int)(i % HD_);
    const size_t t = i / HD_;
    const int p = (int)(t % SP_);
    const int bh = (int)(t / SP_);
    const int b = bh / H_, h = bh % H_;
    float qv = 0.f, kv = 0.f, vv = 0.f;
    if (p < S_) {
      const size_t base = ((size_t)(b * S_ + p)) * TD_ + h * HD_ + hd;
      qv = qkv[base];
      kv = qkv[base + D_];
      vv = qkv[base + 2 * D_];
    }
    if (p < QP_) q[((size_t)bh * QP_ + p) * HD_ + hd] = (bf16)qv;
    k[((size_t)bh * SP_ + p) * HD_ + hd] = (bf16)kv;
    vt[((size_t)bh * HD_ + hd) * SP_ + p] = (bf16)vv;
  }
}

__global__ void __launch_bounds__(256)
scores_kernel(const float* __restrict__ x, float* __restrict__ sc) {
  __shared__ float red[256];
  const int row = blockIdx.x;
  const float* xr = x + (size_t)row * D_;
  const int tid = threadIdx.x;
  float s = 0.f;
  for (int d = tid; d < D_; d += 256) {
    const float v = xr[d];
    s += v * v;
  }
  red[tid] = s;
  __syncthreads();
  for (int st = 128; st > 0; st >>= 1) {
    if (tid < st) red[tid] += red[tid + st];
    __syncthreads();
  }
  if (tid == 0) sc[row] = sqrtf(red[0]);
}

// top-k (k=98) per batch by rank counting; mask[b,0] forced true.
__global__ void __launch_bounds__(256)
mask_kernel(const float* __restrict__ sc, unsigned char* __restrict__ mask) {
  __shared__ float s[S_];
  const int b = blockIdx.x;
  const int t = threadIdx.x;
  for (int i = t; i < S_; i += 256) s[i] = sc[(size_t)b * S_ + i];
  __syncthreads();
  if (t < S_) {
    const float v = s[t];
    int rank = 0;
    for (int j = 0; j < S_; ++j) {
      const float u = s[j];
      rank += (u > v) || (u == v && j < t);
    }
    mask[(size_t)b * S_ + t] = (unsigned char)((rank < NKEEP_) || (t == 0));
  }
}

// ---------------------------------------------------------------------------
// Host orchestration
// ---------------------------------------------------------------------------
extern "C" void kernel_launch(void* const* d_in, const int* in_sizes, int n_in,
                              void* d_out, int out_size, void* d_ws,
                              size_t ws_size, hipStream_t stream) {
  (void)in_sizes; (void)n_in; (void)out_size; (void)ws_size;
  const float* images  = (const float*)d_in[0];
  const float* patch_w = (const float*)d_in[1];
  const float* patch_b = (const float*)d_in[2];
  const float* cls_tok = (const float*)d_in[3];
  const float* pos     = (const float*)d_in[4];
  const float* ln1_w   = (const float*)d_in[5];
  const float* ln1_b   = (const float*)d_in[6];
  const float* qkv_w   = (const float*)d_in[7];
  const float* qkv_b   = (const float*)d_in[8];
  const float* proj_w  = (const float*)d_in[9];
  const float* proj_b  = (const float*)d_in[10];
  const float* ln2_w   = (const float*)d_in[11];
  const float* ln2_b   = (const float*)d_in[12];
  const float* fc1_w   = (const float*)d_in[13];
  const float* fc1_b   = (const float*)d_in[14];
  const float* fc2_w   = (const float*)d_in[15];
  const float* fc2_b   = (const float*)d_in[16];
  const float* norm_w  = (const float*)d_in[17];
  const float* norm_b  = (const float*)d_in[18];
  const float* head_w  = (const float*)d_in[19];
  const float* head_b  = (const float*)d_in[20];

  char* ws = (char*)d_ws;
  size_t off = 0;
  auto alloc = [&](size_t bytes) -> char* {
    char* p = ws + off;
    off += (bytes + 255) & ~(size_t)255;
    return p;
  };

  bf16* wqkv   = (bf16*)alloc((size_t)L_ * TD_ * D_ * 2);
  bf16* wproj  = (bf16*)alloc((size_t)L_ * D_ * D_ * 2);
  bf16* wfc1   = (bf16*)alloc((size_t)L_ * MLP_ * D_ * 2);
  bf16* wfc2   = (bf16*)alloc((size_t)L_ * D_ * MLP_ * 2);
  bf16* wpatch = (bf16*)alloc((size_t)D_ * D_ * 2);
  bf16* whead  = (bf16*)alloc((size_t)1000 * D_ * 2);
  float* x     = (float*)alloc((size_t)MTOK_ * D_ * 4);
  bf16* xn     = (bf16*)alloc((size_t)MTOK_ * D_ * 2);   // aliases patches
  float* qkvf  = (float*)alloc((size_t)MTOK_ * TD_ * 4); // aliases patch-embed out
  bf16* qb     = (bf16*)alloc((size_t)BH_ * QP_ * HD_ * 2);
  bf16* kb     = (bf16*)alloc((size_t)BH_ * SP_ * HD_ * 2);
  bf16* vtb    = (bf16*)alloc((size_t)BH_ * HD_ * SP_ * 2);
  bf16* ob     = (bf16*)alloc((size_t)MTOK_ * D_ * 2);
  bf16* h1     = (bf16*)alloc((size_t)MTOK_ * MLP_ * 2);
  float* sc    = (float*)alloc((size_t)MTOK_ * 4);
  unsigned char* mk = (unsigned char*)alloc((size_t)MTOK_);
  bf16* clsb   = (bf16*)alloc((size_t)B_ * D_ * 2);

  auto nblk = [](size_t n) -> int {
    size_t b = (n + 255) / 256;
    return (int)(b > 262144 ? 262144 : b);
  };
  auto cvt = [&](const float* s, bf16* d, size_t n) {
    cvt_kernel<<<nblk(n), 256, 0, stream>>>(s, d, n);
  };
  auto gemm = [&](const bf16* A, const bf16* W, int M, int N, int K,
                  const float* bias, const float* resid, float* oF, bf16* oB,
                  int g) {
    dim3 grid((N + 127) / 128, (M + 127) / 128);
    gemm_bf16_kernel<<<grid, 128, 0, stream>>>(A, W, M, N, K, bias, resid, oF,
                                               oB, g);
  };

  // 1) one-time (per launch) weight conversion to bf16 (L2-resident afterwards)
  cvt(qkv_w, wqkv, (size_t)L_ * TD_ * D_);
  cvt(proj_w, wproj, (size_t)L_ * D_ * D_);
  cvt(fc1_w, wfc1, (size_t)L_ * MLP_ * D_);
  cvt(fc2_w, wfc2, (size_t)L_ * D_ * MLP_);
  cvt(patch_w, wpatch, (size_t)D_ * D_);
  cvt(head_w, whead, (size_t)1000 * D_);

  // 2) patch embedding: im2col + GEMM + assemble (cls + pos)
  im2col_kernel<<<nblk((size_t)MPATCH_ * D_), 256, 0, stream>>>(images, xn);
  gemm(xn, wpatch, MPATCH_, D_, D_, patch_b, nullptr, qkvf, nullptr, 0);
  assemble_kernel<<<nblk((size_t)MTOK_ * D_), 256, 0, stream>>>(qkvf, cls_tok,
                                                                pos, x);

  // 3) transformer blocks
  for (int i = 0; i < L_; ++i) {
    layernorm_kernel<<<MTOK_, 256, 0, stream>>>(x, D_, ln1_w + (size_t)i * D_,
                                                ln1_b + (size_t)i * D_, xn,
                                                MTOK_);
    gemm(xn, wqkv + (size_t)i * TD_ * D_, MTOK_, TD_, D_,
         qkv_b + (size_t)i * TD_, nullptr, qkvf, nullptr, 0);
    repack_kernel<<<nblk((size_t)BH_ * SP_ * HD_), 256, 0, stream>>>(qkvf, qb,
                                                                     kb, vtb);
    const unsigned char* m = (i >= 4) ? mk : nullptr;
    attention_kernel<<<BH_, 256, ATTN_LDS, stream>>>(qb, kb, vtb, m, ob);
    gemm(ob, wproj + (size_t)i * D_ * D_, MTOK_, D_, D_,
         proj_b + (size_t)i * D_, x, x, nullptr, 0);
    layernorm_kernel<<<MTOK_, 256, 0, stream>>>(x, D_, ln2_w + (size_t)i * D_,
                                                ln2_b + (size_t)i * D_, xn,
                                                MTOK_);
    gemm(xn, wfc1 + (size_t)i * MLP_ * D_, MTOK_, MLP_, D_,
         fc1_b + (size_t)i * MLP_, nullptr, nullptr, h1, 1);
    gemm(h1, wfc2 + (size_t)i * D_ * MLP_, MTOK_, D_, MLP_,
         fc2_b + (size_t)i * D_, x, x, nullptr, 0);
    if (i == 3) {  // token pruning after block 4
      scores_kernel<<<MTOK_, 256, 0, stream>>>(x, sc);
      mask_kernel<<<B_, 256, 0, stream>>>(sc, mk);
    }
  }

  // 4) final LN on cls token + classification head
  layernorm_kernel<<<B_, 256, 0, stream>>>(x, (long long)S_ * D_, norm_w,
                                           norm_b, clsb, B_);
  gemm(clsb, whead, B_, 1000, D_, head_b, nullptr, (float*)d_out, nullptr, 0);
}
